// RPN_67061619360385
// MI455X (gfx1250) — matrix operand order repaired and checked by reading, not compiled, not run on hardware
//
#include <hip/hip_runtime.h>
#include <math.h>

// ---------------------------------------------------------------------------
// RPN for MI455X (gfx1250, wave32). Stem 3x3 conv = implicit GEMM via
// v_wmma_f32_16x16x32_bf16 (M=B*H*W, N=256, K=2304), fused 1x1 heads,
// then a per-(image,level) top-k/decode/NMS kernel.
//  - fragment-ready bf16 weight packing -> B staging is a b128 memcpy
//  - B staging uses GLOBAL_LOAD_ASYNC_TO_LDS_B128 (ASYNCcnt-tracked DMA)
//  - ping-pong LDS double buffering, 1 barrier per K-step
//  - clamped-address + cndmask boundary handling (no exec divergence)
//  - 64M x 256N block tile: 8 WMMAs per A fragment
// ---------------------------------------------------------------------------

typedef __attribute__((ext_vector_type(16))) __bf16 v16bf;
typedef __attribute__((ext_vector_type(8)))  float  v8f;
typedef int v4i __attribute__((vector_size(16)));            // builtin's elt type
typedef __attribute__((address_space(1))) v4i* as1_v4i;       // global int4*
typedef __attribute__((address_space(3))) v4i* as3_v4i;       // lds int4*

#define C_IN     256
#define BATCH    8
#define APITCH   40   // halfs/row: 32 data + 8 pad (80B: 16B-aligned, conflict-free)
#define BPITCH   40
#define SPITCH   264  // floats/row for activation buffer
#define PRE_TOPK 400
#define POST_TOPK 50

#if defined(__HIP_DEVICE_COMPILE__) && \
    __has_builtin(__builtin_amdgcn_global_load_async_to_lds_b128)
#define HAVE_ASYNC_LDS 1
#else
#define HAVE_ASYNC_LDS 0
#endif

static __device__ __forceinline__ void wait_async_lds() {
#if HAVE_ASYNC_LDS
#if __has_builtin(__builtin_amdgcn_s_wait_asynccnt)
    __builtin_amdgcn_s_wait_asynccnt(0);
#else
    asm volatile("s_wait_asynccnt 0x0" ::: "memory");
#endif
#endif
}

// ---- workspace layout (bytes) ----
constexpr size_t BW_OFF   = 0;
constexpr size_t BW_SZ    = 72ull * 256 * 32 * 2;         // frag-ready bf16 weights
constexpr size_t OBJ3_OFF = BW_OFF + BW_SZ;
constexpr size_t OBJ3_SZ  = 8ull * 128 * 128 * 3 * 4;
constexpr size_t OBJ4_OFF = OBJ3_OFF + OBJ3_SZ;
constexpr size_t OBJ4_SZ  = 8ull * 64 * 64 * 3 * 4;
constexpr size_t OBJ5_OFF = OBJ4_OFF + OBJ4_SZ;
constexpr size_t OBJ5_SZ  = 8ull * 32 * 32 * 3 * 4;
constexpr size_t BOX3_OFF = OBJ5_OFF + OBJ5_SZ;
constexpr size_t BOX3_SZ  = OBJ3_SZ * 4;
constexpr size_t BOX4_OFF = BOX3_OFF + BOX3_SZ;
constexpr size_t BOX4_SZ  = OBJ4_SZ * 4;
constexpr size_t BOX5_OFF = BOX4_OFF + BOX4_SZ;

static __device__ __forceinline__ unsigned short f2bf(float f) {
    unsigned u = __float_as_uint(f);
    unsigned r = u + 0x7fffu + ((u >> 16) & 1u);   // round-to-nearest-even
    return (unsigned short)(r >> 16);
}

// ---------------------------------------------------------------------------
// Kernel 0: repack w_stem (co,ci,3,3) f32 -> bf16, fragment-ready layout
//   bw2[kb][n][k] ,  K = (ky*3+kx)*256 + ci ,  kb = K>>5 , k = K&31 , n = co
// ---------------------------------------------------------------------------
__global__ void rpn_wcvt(const float* __restrict__ w_stem,
                         unsigned short* __restrict__ bw2) {
    int K  = blockIdx.x;          // 0..2303
    int co = threadIdx.x;         // 0..255
    int tap = K >> 8, ci = K & 255;
    int ky = tap / 3, kx = tap % 3;
    float v = w_stem[((co * C_IN + ci) * 3 + ky) * 3 + kx];
    bw2[(size_t)(K >> 5) * 8192 + co * 32 + (K & 31)] = f2bf(v);
}

// ---------------------------------------------------------------------------
// Kernel 1: fused stem GEMM + ReLU + 1x1 heads.
// Block = 256 threads = 8 waves as 4(M) x 2(N); block tile 64M x 256N.
// K loop: 72 steps of 32 (9 taps x 8 ci-blocks), ping-pong staged.
// ---------------------------------------------------------------------------
struct SharedMem {
    union {
        struct {
            unsigned short A[2][64 * APITCH];    //  10.2 KB
            unsigned short B[2][256 * BPITCH];   //  41.0 KB
        } g;
        float S[32 * SPITCH];                    //  33.8 KB (epilogue only)
    };
};

__global__ void __launch_bounds__(256)
rpn_stem_heads(const float* __restrict__ feat,          // (8,256,H,W)
               const unsigned short* __restrict__ bw2,  // (72,256,32) bf16
               const float* __restrict__ b_stem,
               const float* __restrict__ w_obj, const float* __restrict__ b_obj,
               const float* __restrict__ w_box, const float* __restrict__ b_box,
               float* __restrict__ objw,                // (8, HW*3)
               float* __restrict__ boxw,                // (8, HW*3, 4)
               int H, int W) {
    __shared__ __align__(16) SharedMem sh;

    const int tid  = threadIdx.x;
    const int lane = tid & 31;
    const int wave = tid >> 5;      // 0..7
    const int wm   = wave >> 1;     // 0..3 : wave M row (16 rows each)
    const int wn   = wave & 1;      // 0..1 : wave N col (128 cols each)
    const int hi   = lane >> 4;
    const int l15  = lane & 15;
    const int HW   = H * W;
    const int mbase = blockIdx.x * 64;

    v8f acc[8];
    v8f zero = {0.f, 0.f, 0.f, 0.f, 0.f, 0.f, 0.f, 0.f};
#pragma unroll
    for (int j = 0; j < 8; ++j) acc[j] = zero;

    // ---- per-thread A-staging coordinates (row am, 8 consecutive ci) ----
    const int am  = tid & 63;
    const int kk0 = (tid >> 6) * 8;          // 0,8,16,24
    const int gm  = mbase + am;
    const int img = gm / HW;
    const int pos = gm - img * HW;
    const int ay  = pos / W;
    const int ax  = pos - ay * W;
    const float* fbase = feat + (size_t)img * C_IN * HW;

    auto stageA = [&](int kit, int buf) {
        int tap = kit >> 3, cb = kit & 7;
        int ky = tap / 3, kx = tap - ky * 3;
        int iy = ay + ky - 1, ix = ax + kx - 1;
        bool inb = ((unsigned)iy < (unsigned)H) && ((unsigned)ix < (unsigned)W);
        int cy = iy < 0 ? 0 : (iy >= H ? H - 1 : iy);   // clamp: always-legal load
        int cx = ix < 0 ? 0 : (ix >= W ? W - 1 : ix);
        const float* fpix = fbase + (size_t)cy * W + cx + (size_t)(cb * 32 + kk0) * HW;
        unsigned short* dst = sh.g.A[buf] + am * APITCH + kk0;
#pragma unroll
        for (int r = 0; r < 4; ++r) {
            float v0 = fpix[(size_t)(2 * r) * HW];
            float v1 = fpix[(size_t)(2 * r + 1) * HW];
            v0 = inb ? v0 : 0.f;            // v_cndmask, no exec branching
            v1 = inb ? v1 : 0.f;
            unsigned pk = (unsigned)f2bf(v0) | ((unsigned)f2bf(v1) << 16);
            *(unsigned*)(dst + 2 * r) = pk;
        }
    };

    auto stageB = [&](int kit, int buf) {
        // one full fragment-ready row (64B) per thread
        const uint4* src = (const uint4*)(bw2 + (size_t)kit * 8192) + tid * 4;
        __builtin_prefetch(src + 1024, 0, 1);           // next K-block (L2 hint)
        unsigned short* dst = sh.g.B[buf] + tid * BPITCH;
#if HAVE_ASYNC_LDS
#pragma unroll
        for (int r = 0; r < 4; ++r)
            __builtin_amdgcn_global_load_async_to_lds_b128(
                (as1_v4i)(uint4*)(src + r),          // global src (AS1, non-const)
                (as3_v4i)(void*)(dst + r * 8),       // lds dst (AS3)
                0, 0);
#else
        uint4 q0 = src[0], q1 = src[1], q2 = src[2], q3 = src[3];
        *(uint4*)(dst)      = q0;
        *(uint4*)(dst + 8)  = q1;
        *(uint4*)(dst + 16) = q2;
        *(uint4*)(dst + 24) = q3;
#endif
    };

    stageA(0, 0);
    stageB(0, 0);
    wait_async_lds();
    __syncthreads();

    for (int kit = 0; kit < 72; ++kit) {
        const int cur = kit & 1;
        if (kit + 1 < 72) {                 // uniform branch
            stageA(kit + 1, cur ^ 1);
            stageB(kit + 1, cur ^ 1);
        }
        // A fragment: 2 x b128 (16-bit A 16x32 layout, lane-half k-interleave)
        union { uint4 q[2]; v16bf v; } af;
        const unsigned short* arow = sh.g.A[cur] + (wm * 16 + l15) * APITCH;
        af.q[0] = *(const uint4*)(arow + hi * 8);
        af.q[1] = *(const uint4*)(arow + 16 + hi * 8);
#pragma unroll
        for (int j = 0; j < 8; ++j) {
            union { uint4 q[2]; v16bf v; } bf;
            const unsigned short* brow =
                sh.g.B[cur] + (wn * 128 + j * 16 + l15) * BPITCH;
            bf.q[0] = *(const uint4*)(brow + hi * 8);
            bf.q[1] = *(const uint4*)(brow + 16 + hi * 8);
            acc[j] = __builtin_amdgcn_wmma_f32_16x16x32_bf16(
                false, af.v, false, bf.v, (short)0, acc[j], false, false);
        }
        wait_async_lds();   // drain this wave's async staging before publishing
        __syncthreads();
    }

    // ---- epilogue: bias+ReLU -> LDS (f32), heads; two 32-row phases so the
    //      activation buffer can overlay the (now dead) staging buffers.
    for (int ph = 0; ph < 2; ++ph) {
        if ((wm >> 1) == ph) {
#pragma unroll
            for (int j = 0; j < 8; ++j) {
                int n = wn * 128 + j * 16 + l15;
                float bias = b_stem[n];
#pragma unroll
                for (int r = 0; r < 8; ++r) {
                    int m = (wm & 1) * 16 + hi * 8 + r;     // 0..31 in phase
                    float s = acc[j][r] + bias;
                    sh.S[m * SPITCH + n] = s > 0.f ? s : 0.f;
                }
            }
        }
        __syncthreads();

        for (int t = tid; t < 32 * 15; t += 256) {
            int m = t / 15, o = t % 15;
            const float* wrow = (o < 3) ? (w_obj + o * C_IN)
                                        : (w_box + (o - 3) * C_IN);
            float sum = (o < 3) ? b_obj[o] : b_box[o - 3];
            const float* srow = sh.S + m * SPITCH;
#pragma unroll 8
            for (int c = 0; c < C_IN; ++c) sum += srow[c] * wrow[c];
            int gmo = mbase + ph * 32 + m;
            int b   = gmo / HW;
            int p   = gmo - b * HW;
            if (o < 3) {
                objw[(size_t)b * HW * 3 + p * 3 + o] = sum;
            } else {
                int ch = o - 3, a = ch >> 2, cc = ch & 3;
                boxw[((size_t)b * HW * 3 + p * 3 + a) * 4 + cc] = sum;
            }
        }
        __syncthreads();
    }
}

// ---------------------------------------------------------------------------
// Kernel 2: per-(image,level) proposals: top-400, decode, NMS, top-50.
// ---------------------------------------------------------------------------
__global__ void __launch_bounds__(256)
rpn_proposals(char* __restrict__ ws, float* __restrict__ out) {
    const int lvl = blockIdx.x;
    const int b   = blockIdx.y;
    const int tid = threadIdx.x;

    const int    Wtab[3]   = {128, 64, 32};
    const float  stab[3]   = {8.f, 16.f, 32.f};
    const size_t objOff[3] = {OBJ3_OFF, OBJ4_OFF, OBJ5_OFF};
    const size_t boxOff[3] = {BOX3_OFF, BOX4_OFF, BOX5_OFF};

    const int W = Wtab[lvl];
    const int N = W * W * 3;
    float*       scores = (float*)(ws + objOff[lvl]) + (size_t)b * N;
    const float* deltas = (float*)(ws + boxOff[lvl]) + (size_t)b * N * 4;

    __shared__ float topS[PRE_TOPK];
    __shared__ int   topI[PRE_TOPK];
    __shared__ float bx[PRE_TOPK][4];
    __shared__ float area[PRE_TOPK];
    __shared__ int   keep[PRE_TOPK];
    __shared__ float red_s[256];
    __shared__ int   red_i[256];
    __shared__ int   sel[POST_TOPK];

    // ---- iterative block-argmax top-400 (scores stay L2-resident)
    for (int t = 0; t < PRE_TOPK; ++t) {
        float best = -INFINITY;
        int   bi   = 0x7fffffff;
        for (int i = tid; i < N; i += 256) {
            float v = scores[i];
            if (v > best) { best = v; bi = i; }
        }
        red_s[tid] = best; red_i[tid] = bi;
        __syncthreads();
        for (int s = 128; s > 0; s >>= 1) {
            if (tid < s) {
                bool take = (red_s[tid + s] > red_s[tid]) ||
                            (red_s[tid + s] == red_s[tid] &&
                             red_i[tid + s] < red_i[tid]);
                if (take) { red_s[tid] = red_s[tid + s]; red_i[tid] = red_i[tid + s]; }
            }
            __syncthreads();
        }
        if (tid == 0) {
            topS[t] = red_s[0];
            topI[t] = red_i[0];
            scores[red_i[0]] = -INFINITY;   // recomputed next launch
        }
        __syncthreads();
    }

    // ---- decode (anchors on the fly)
    const float ratios[3] = {0.5f, 1.f, 2.f};
    for (int t = tid; t < PRE_TOPK; t += 256) {
        int idx = topI[t];
        int p = idx / 3, a = idx - p * 3;
        int y = p / W, x = p - y * W;
        float stride = stab[lvl];
        float cx = (x + 0.5f) * stride, cy = (y + 0.5f) * stride;
        float size = stride * 8.f;
        float sq = sqrtf(ratios[a]);
        float aw = size / sq, ah = size * sq;
        const float* d = deltas + (size_t)idx * 4;
        const float clampv = 4.135166556742356f;
        float dx = d[0], dy = d[1];
        float dw = fminf(fmaxf(d[2], -clampv), clampv);
        float dh = fminf(fmaxf(d[3], -clampv), clampv);
        float pcx = cx + dx * aw, pcy = cy + dy * ah;
        float pw = aw * expf(dw), ph = ah * expf(dh);
        float x1 = pcx - 0.5f * pw, y1 = pcy - 0.5f * ph;
        float x2 = pcx + 0.5f * pw, y2 = pcy + 0.5f * ph;
        bx[t][0] = x1; bx[t][1] = y1; bx[t][2] = x2; bx[t][3] = y2;
        area[t]  = fmaxf(x2 - x1, 0.f) * fmaxf(y2 - y1, 0.f);
        keep[t]  = 1;
    }
    __syncthreads();

    // ---- greedy NMS, parallel suppression
    for (int i = 0; i < PRE_TOPK; ++i) {
        if (keep[i]) {   // uniform (LDS, synced)
            float x1 = bx[i][0], y1 = bx[i][1], x2 = bx[i][2], y2 = bx[i][3];
            float ai = area[i];
            for (int j = i + 1 + tid; j < PRE_TOPK; j += 256) {
                if (keep[j]) {
                    float ix1 = fmaxf(x1, bx[j][0]);
                    float iy1 = fmaxf(y1, bx[j][1]);
                    float ix2 = fminf(x2, bx[j][2]);
                    float iy2 = fminf(y2, bx[j][3]);
                    float inter = fmaxf(ix2 - ix1, 0.f) * fmaxf(iy2 - iy1, 0.f);
                    float iou = inter / fmaxf(ai + area[j] - inter, 1e-8f);
                    if (iou > 0.6f) keep[j] = 0;
                }
            }
        }
        __syncthreads();
    }

    // ---- top-50 of masked scores (kept in order, then suppressed by index)
    if (tid == 0) {
        int c = 0;
        for (int t = 0; t < PRE_TOPK && c < POST_TOPK; ++t) if (keep[t])  sel[c++] = t;
        for (int t = 0; t < PRE_TOPK && c < POST_TOPK; ++t) if (!keep[t]) sel[c++] = t;
    }
    __syncthreads();

    for (int t = tid; t < POST_TOPK * 4; t += 256) {
        int i = t >> 2, c = t & 3;
        out[((size_t)b * 150 + lvl * 50 + i) * 4 + c] = bx[sel[i]][c];
    }
}

// ---------------------------------------------------------------------------
extern "C" void kernel_launch(void* const* d_in, const int* in_sizes, int n_in,
                              void* d_out, int out_size, void* d_ws, size_t ws_size,
                              hipStream_t stream) {
    const float* p3     = (const float*)d_in[0];
    const float* p4     = (const float*)d_in[1];
    const float* p5     = (const float*)d_in[2];
    const float* w_stem = (const float*)d_in[3];
    const float* b_stem = (const float*)d_in[4];
    const float* w_obj  = (const float*)d_in[5];
    const float* b_obj  = (const float*)d_in[6];
    const float* w_box  = (const float*)d_in[7];
    const float* b_box  = (const float*)d_in[8];
    float* out = (float*)d_out;

    char* ws = (char*)d_ws;
    unsigned short* bw2 = (unsigned short*)(ws + BW_OFF);

    // 0) repack weights to fragment-ready bf16 layout
    rpn_wcvt<<<2304, 256, 0, stream>>>(w_stem, bw2);

    // 1) fused stem + heads, one launch per level (64-row M tiles)
    rpn_stem_heads<<<BATCH * 128 * 128 / 64, 256, 0, stream>>>(
        p3, bw2, b_stem, w_obj, b_obj, w_box, b_box,
        (float*)(ws + OBJ3_OFF), (float*)(ws + BOX3_OFF), 128, 128);
    rpn_stem_heads<<<BATCH * 64 * 64 / 64, 256, 0, stream>>>(
        p4, bw2, b_stem, w_obj, b_obj, w_box, b_box,
        (float*)(ws + OBJ4_OFF), (float*)(ws + BOX4_OFF), 64, 64);
    rpn_stem_heads<<<BATCH * 32 * 32 / 64, 256, 0, stream>>>(
        p5, bw2, b_stem, w_obj, b_obj, w_box, b_box,
        (float*)(ws + OBJ5_OFF), (float*)(ws + BOX5_OFF), 32, 32);

    // 2) proposals
    rpn_proposals<<<dim3(3, BATCH), 256, 0, stream>>>(ws, out);
}
